// Attention_39788577030373
// MI455X (gfx1250) — matrix operand order repaired and checked by reading, not compiled, not run on hardware
//
#include <hip/hip_runtime.h>
#include <hip/hip_bf16.h>

typedef _Float16 f16;
typedef __attribute__((ext_vector_type(16))) _Float16 v16h;
typedef __attribute__((ext_vector_type(8)))  _Float16 v8h;
typedef __attribute__((ext_vector_type(4)))  _Float16 v4h;
typedef __attribute__((ext_vector_type(8)))  float    v8f;

#define D_MODEL 1024
#define SEQ     1024
#define BATCH   8
#define HEADS   16
#define DHEAD   64
#define BS      (BATCH * SEQ)        // 8192 rows
#define QKV_N   (3 * D_MODEL)        // 3072

// ---------------------------------------------------------------- helpers

__device__ inline v8f vzero8() {
  v8f z = {0.f, 0.f, 0.f, 0.f, 0.f, 0.f, 0.f, 0.f};
  return z;
}

// low 32 bits of a generic pointer into LDS == group-relative LDS byte address
__device__ inline uint32_t lds_addr32(const void* p) {
  return (uint32_t)(uintptr_t)p;
}

// CDNA5 async DMA: global (SGPR base + 32-bit VGPR offset) -> LDS, 16B per lane.
// Tracked by ASYNCcnt; drained with s_wait_asynccnt.
__device__ inline void async_load_b128(uint32_t lds_byte, uint32_t goff, uint64_t gbase) {
  asm volatile("global_load_async_to_lds_b128 %0, %1, %2"
               :: "v"(lds_byte), "v"(goff), "s"(gbase)
               : "memory");
}
__device__ inline void wait_async0() { asm volatile("s_wait_asynccnt 0x0" ::: "memory"); }
__device__ inline void wait_async2() { asm volatile("s_wait_asynccnt 0x2" ::: "memory"); }
__device__ inline void wait_async3() { asm volatile("s_wait_asynccnt 0x3" ::: "memory"); }

// A fragment: 16x32 f16, rows m=lane&15, per-lane halves hold K 0..15 / 16..31 (contiguous)
__device__ inline v16h load_frag_a(const f16* base, int ld, int lane) {
  const int r  = lane & 15;
  const int lh = (lane >> 4) & 1;
  const f16* p = base + (size_t)r * ld;
  v8h lo = *(const v8h*)(p + lh * 16);
  v8h hi = *(const v8h*)(p + lh * 16 + 8);
  v16h out;
#pragma unroll
  for (int i = 0; i < 8; ++i) { out[i] = lo[i]; out[8 + i] = hi[i]; }
  return out;
}

// B fragment: 32x16, where B[k][n] = base[n*ld + k]
// lanes 0-15 hold K=0..15, lanes 16-31 hold K=16..31 (contiguous 16 halfs per lane)
__device__ inline v16h load_frag_b_rows(const f16* base, int ld, int lane) {
  const int n  = lane & 15;
  const int lh = (lane >> 4) & 1;
  const f16* p = base + (size_t)n * ld + lh * 16;
  v8h lo = *(const v8h*)(p);
  v8h hi = *(const v8h*)(p + 8);
  v16h out;
#pragma unroll
  for (int i = 0; i < 8; ++i) { out[i] = lo[i]; out[8 + i] = hi[i]; }
  return out;
}

__device__ inline v8f wmma_f16(v16h a, v16h b, v8f c) {
  return __builtin_amdgcn_wmma_f32_16x16x32_f16(false, a, false, b, (short)0, c,
                                                false, false);
}

// 16-lane row reductions (D-layout rows live across lanes 0..15 / 16..31)
__device__ inline float rowmax16(float v) {
  v = fmaxf(v, __shfl_xor(v, 1, 32));
  v = fmaxf(v, __shfl_xor(v, 2, 32));
  v = fmaxf(v, __shfl_xor(v, 4, 32));
  v = fmaxf(v, __shfl_xor(v, 8, 32));
  return v;
}
__device__ inline float rowsum16(float v) {
  v += __shfl_xor(v, 1, 32);
  v += __shfl_xor(v, 2, 32);
  v += __shfl_xor(v, 4, 32);
  v += __shfl_xor(v, 8, 32);
  return v;
}

// Shared async-pipelined GEMM mainloop.
// Block tile: 128(M) x 64(N), K stepped by 32, double-buffered LDS.
// 8 waves: wave (wm = wave&3, wn = wave>>2) computes 32x32 via 4 accumulators.
__device__ inline void gemm_mainloop(const f16* Abase, const f16* Bbase, int ld,
                                     f16* ldsA, f16* ldsB,
                                     int t, int lane, int wave, v8f acc[4]) {
  const uint32_t ldsAb = lds_addr32(ldsA);
  const uint32_t ldsBb = lds_addr32(ldsB);
  const uint64_t gA = (uint64_t)(uintptr_t)Abase;
  const uint64_t gB = (uint64_t)(uintptr_t)Bbase;
  // per-thread global byte offsets for stage k0=0
  const uint32_t offA = ((uint32_t)(t >> 1) * ld + (uint32_t)(t & 1) * 16) * 2;
  const uint32_t offB = ((uint32_t)(t >> 2) * ld + (uint32_t)(t & 3) * 8) * 2;
  const uint32_t ldsA_t = (uint32_t)t * 32;    // row-major 128x32 tile, 16B chunks
  const uint32_t ldsB_t = (uint32_t)t * 16;    // row-major  64x32 tile
  const int wm = wave & 3, wn = wave >> 2;

  // prologue: stage 0 into buffer 0
  async_load_b128(ldsAb + ldsA_t,      offA,      gA);
  async_load_b128(ldsAb + ldsA_t + 16, offA + 16, gA);
  async_load_b128(ldsBb + ldsB_t,      offB,      gB);

  for (int j = 0; j < D_MODEL / 32; ++j) {
    const int buf = j & 1;
    if (j + 1 < D_MODEL / 32) {
      const uint32_t kb = (uint32_t)(j + 1) * 64;        // 32 halfs = 64B along K
      const uint32_t lA = ldsAb + (uint32_t)(buf ^ 1) * (128 * 32 * 2) + ldsA_t;
      const uint32_t lB = ldsBb + (uint32_t)(buf ^ 1) * (64 * 32 * 2) + ldsB_t;
      async_load_b128(lA,      offA + kb,      gA);
      async_load_b128(lA + 16, offA + kb + 16, gA);
      async_load_b128(lB,      offB + kb,      gB);
      wait_async3();                 // stage j complete, stage j+1 in flight
    } else {
      wait_async0();
    }
    __syncthreads();

    const f16* As = ldsA + (size_t)buf * (128 * 32) + (wm * 32) * 32;
    const f16* Bs = ldsB + (size_t)buf * (64 * 32) + (wn * 32) * 32;
    const v16h a0 = load_frag_a(As, 32, lane);
    const v16h a1 = load_frag_a(As + 16 * 32, 32, lane);
    const v16h b0 = load_frag_b_rows(Bs, 32, lane);
    const v16h b1 = load_frag_b_rows(Bs + 16 * 32, 32, lane);
    acc[0] = wmma_f16(a0, b0, acc[0]);
    acc[1] = wmma_f16(a0, b1, acc[1]);
    acc[2] = wmma_f16(a1, b0, acc[2]);
    acc[3] = wmma_f16(a1, b1, acc[3]);
    __syncthreads();
  }
}

// ---------------------------------------------------------------- kernels

// f32 -> f16 conversion, 4 elements/thread
__global__ __launch_bounds__(256) void cvt_f32_to_f16(const float* __restrict__ in,
                                                      f16* __restrict__ out, int n4) {
  int i = blockIdx.x * 256 + threadIdx.x;
  if (i >= n4) return;
  const float4 v = *(const float4*)(in + (size_t)i * 4);
  v4h o;
  o[0] = (f16)v.x; o[1] = (f16)v.y; o[2] = (f16)v.z; o[3] = (f16)v.w;
  *(v4h*)(out + (size_t)i * 4) = o;
}

// LayerNorm: one block per row of 1024, output f16
__global__ __launch_bounds__(256) void ln_kernel(const float* __restrict__ x,
                                                 const float* __restrict__ gamma,
                                                 const float* __restrict__ beta,
                                                 f16* __restrict__ xn) {
  const int row  = blockIdx.x;
  const int lane = threadIdx.x & 31;
  const int wave = threadIdx.x >> 5;
  const float* xr = x + (size_t)row * D_MODEL;
  const int c0 = threadIdx.x * 4;
  const float4 v = *(const float4*)(xr + c0);

  float s  = v.x + v.y + v.z + v.w;
  float sq = v.x * v.x + v.y * v.y + v.z * v.z + v.w * v.w;
#pragma unroll
  for (int m = 16; m > 0; m >>= 1) {
    s  += __shfl_xor(s, m, 32);
    sq += __shfl_xor(sq, m, 32);
  }
  __shared__ float sh_s[8], sh_q[8];
  if (lane == 0) { sh_s[wave] = s; sh_q[wave] = sq; }
  __syncthreads();
  float ts = 0.f, tq = 0.f;
#pragma unroll
  for (int w = 0; w < 8; ++w) { ts += sh_s[w]; tq += sh_q[w]; }
  const float mean = ts * (1.f / D_MODEL);
  const float var  = tq * (1.f / D_MODEL) - mean * mean;
  const float rstd = rsqrtf(var + 1e-5f);

  const float4 g  = *(const float4*)(gamma + c0);
  const float4 bb = *(const float4*)(beta + c0);
  v4h o;
  o[0] = (f16)((v.x - mean) * rstd * g.x + bb.x);
  o[1] = (f16)((v.y - mean) * rstd * g.y + bb.y);
  o[2] = (f16)((v.z - mean) * rstd * g.z + bb.z);
  o[3] = (f16)((v.w - mean) * rstd * g.w + bb.w);
  *(v4h*)(xn + (size_t)row * D_MODEL + c0) = o;
}

// QKV GEMM: qkv[m,e] = sum_d xn[m,d]*W[e,d] + bias[e]
__global__ __launch_bounds__(256) void gemm_qkv(const f16* __restrict__ xnh,
                                                const f16* __restrict__ wh,
                                                const float* __restrict__ bias,
                                                f16* __restrict__ qkvh) {
  __shared__ f16 sA[2 * 128 * 32];
  __shared__ f16 sB[2 * 64 * 32];
  const int t = threadIdx.x;
  const int lane = t & 31;
  const int wave = t >> 5;
  const int tilesN = QKV_N / 64;                // 48
  const int m0 = (blockIdx.x / tilesN) * 128;
  const int n0 = (blockIdx.x % tilesN) * 64;

  v8f acc[4];
#pragma unroll
  for (int i = 0; i < 4; ++i) acc[i] = vzero8();

  gemm_mainloop(xnh + (size_t)m0 * D_MODEL, wh + (size_t)n0 * D_MODEL, D_MODEL,
                sA, sB, t, lane, wave, acc);

  const int wm = wave & 3, wn = wave >> 2;
  const int cl = lane & 15;
  const int hi = (lane >> 4) & 1;
#pragma unroll
  for (int mi = 0; mi < 2; ++mi) {
#pragma unroll
    for (int ni = 0; ni < 2; ++ni) {
      const v8f a = acc[mi * 2 + ni];
      const int rbase = m0 + wm * 32 + mi * 16 + hi * 8;
      const int col = n0 + wn * 32 + ni * 16 + cl;
      const float bi = bias[col];
#pragma unroll
      for (int i = 0; i < 8; ++i)
        qkvh[(size_t)(rbase + i) * QKV_N + col] = (f16)(a[i] + bi);
    }
  }
}

// Flash attention: block = 8 waves = 128 query rows of one (b,h).
// K/V 32-key tiles DMAed into double-buffered LDS with async loads.
__global__ __launch_bounds__(256) void attn_kernel(const f16* __restrict__ qkvh,
                                                   f16* __restrict__ ctxh) {
  __shared__ f16 ldsK[2][32 * 64];
  __shared__ f16 ldsV[2][32 * 64];
  __shared__ f16 ldsP[8][16 * 32];

  const int t = threadIdx.x;
  const int lane = t & 31;
  const int wave = t >> 5;
  const int qc = blockIdx.x & 7;
  const int h  = (blockIdx.x >> 3) & 15;
  const int b  = blockIdx.x >> 7;
  const int q0 = qc * 128 + wave * 16;

  const size_t rowbase = (size_t)(b * SEQ) * QKV_N;
  const f16* Qp = qkvh + rowbase + (size_t)q0 * QKV_N + h * DHEAD;
  const f16* Kp = qkvh + rowbase + D_MODEL + h * DHEAD;
  const f16* Vp = qkvh + rowbase + 2 * D_MODEL + h * DHEAD;

  const v16h qa0 = load_frag_a(Qp, QKV_N, lane);
  const v16h qa1 = load_frag_a(Qp + 32, QKV_N, lane);

  v8f acc[4];
#pragma unroll
  for (int nb = 0; nb < 4; ++nb) acc[nb] = vzero8();
  float mrow[8], lrow[8];
#pragma unroll
  for (int i = 0; i < 8; ++i) { mrow[i] = -1e30f; lrow[i] = 0.f; }

  // async staging setup: each thread moves one 16B chunk of K and of V per stage
  const uint32_t ldsKb = lds_addr32(&ldsK[0][0]);
  const uint32_t ldsVb = lds_addr32(&ldsV[0][0]);
  const uint64_t gK = (uint64_t)(uintptr_t)Kp;
  const uint64_t gV = (uint64_t)(uintptr_t)Vp;
  const uint32_t offKV = ((uint32_t)(t >> 3) * QKV_N + (uint32_t)(t & 7) * 8) * 2;
  const uint32_t lds_t = (uint32_t)t * 16;
  const uint32_t stageB = (uint32_t)(32 * QKV_N * 2);    // bytes per 32-key stage

  async_load_b128(ldsKb + lds_t, offKV, gK);
  async_load_b128(ldsVb + lds_t, offKV, gV);

  for (int j = 0; j < SEQ / 32; ++j) {
    const int buf = j & 1;
    if (j + 1 < SEQ / 32) {
      const uint32_t go = offKV + (uint32_t)(j + 1) * stageB;
      const uint32_t lo = (uint32_t)(buf ^ 1) * (32 * 64 * 2) + lds_t;
      async_load_b128(ldsKb + lo, go, gK);
      async_load_b128(ldsVb + lo, go, gV);
      wait_async2();
    } else {
      wait_async0();
    }
    __syncthreads();

    const f16* Kt = &ldsK[buf][0];
    const f16* Vt = &ldsV[buf][0];

    // scores: s0 = keys [0..15], s1 = keys [16..31] of this tile
    v8f s0 = vzero8(), s1 = vzero8();
    s0 = wmma_f16(qa0, load_frag_b_rows(Kt, 64, lane), s0);
    s0 = wmma_f16(qa1, load_frag_b_rows(Kt + 32, 64, lane), s0);
    s1 = wmma_f16(qa0, load_frag_b_rows(Kt + 16 * 64, 64, lane), s1);
    s1 = wmma_f16(qa1, load_frag_b_rows(Kt + 16 * 64 + 32, 64, lane), s1);

    const int cl = lane & 15;
    const int hi = (lane >> 4) & 1;
#pragma unroll
    for (int i = 0; i < 8; ++i) {
      const float a0 = s0[i] * 0.125f;        // 1/sqrt(64)
      const float a1 = s1[i] * 0.125f;
      const float tm = rowmax16(fmaxf(a0, a1));
      const float nm = fmaxf(mrow[i], tm);
      const float corr = __expf(mrow[i] - nm);
      const float p0 = __expf(a0 - nm);
      const float p1 = __expf(a1 - nm);
      const float rs = rowsum16(p0 + p1);
      lrow[i] = lrow[i] * corr + rs;
      mrow[i] = nm;
      acc[0][i] *= corr; acc[1][i] *= corr;
      acc[2][i] *= corr; acc[3][i] *= corr;
      const int row = i + hi * 8;
      ldsP[wave][row * 32 + cl]      = (f16)p0;
      ldsP[wave][row * 32 + 16 + cl] = (f16)p1;
    }
    __builtin_amdgcn_wave_barrier();

    const v16h pa = load_frag_a(&ldsP[wave][0], 32, lane);
#pragma unroll
    for (int nb = 0; nb < 4; ++nb) {
      const int n  = lane & 15;
      const int lh = (lane >> 4) & 1;
      v16h vb;
#pragma unroll
      for (int k = 0; k < 16; ++k)
        vb[k] = Vt[(lh * 16 + k) * 64 + nb * 16 + n];
      acc[nb] = wmma_f16(pa, vb, acc[nb]);
    }
    __syncthreads();
  }

  const int cl = lane & 15;
  const int hi = (lane >> 4) & 1;
#pragma unroll
  for (int nb = 0; nb < 4; ++nb) {
#pragma unroll
    for (int i = 0; i < 8; ++i) {
      const int row = i + hi * 8;
      const float v = acc[nb][i] / lrow[i];
      ctxh[(size_t)(b * SEQ + q0 + row) * D_MODEL + h * DHEAD + nb * 16 + cl] = (f16)v;
    }
  }
}

// Out-projection GEMM + bias + residual, f32 output
__global__ __launch_bounds__(256) void gemm_out(const f16* __restrict__ ctxh,
                                                const f16* __restrict__ wh,
                                                const float* __restrict__ bias,
                                                const float* __restrict__ resid,
                                                float* __restrict__ out) {
  __shared__ f16 sA[2 * 128 * 32];
  __shared__ f16 sB[2 * 64 * 32];
  const int t = threadIdx.x;
  const int lane = t & 31;
  const int wave = t >> 5;
  const int tilesN = D_MODEL / 64;              // 16
  const int m0 = (blockIdx.x / tilesN) * 128;
  const int n0 = (blockIdx.x % tilesN) * 64;

  v8f acc[4];
#pragma unroll
  for (int i = 0; i < 4; ++i) acc[i] = vzero8();

  gemm_mainloop(ctxh + (size_t)m0 * D_MODEL, wh + (size_t)n0 * D_MODEL, D_MODEL,
                sA, sB, t, lane, wave, acc);

  const int wm = wave & 3, wn = wave >> 2;
  const int cl = lane & 15;
  const int hi = (lane >> 4) & 1;
#pragma unroll
  for (int mi = 0; mi < 2; ++mi) {
#pragma unroll
    for (int ni = 0; ni < 2; ++ni) {
      const v8f a = acc[mi * 2 + ni];
      const int rbase = m0 + wm * 32 + mi * 16 + hi * 8;
      const int col = n0 + wn * 32 + ni * 16 + cl;
      const float bi = bias[col];
#pragma unroll
      for (int i = 0; i < 8; ++i)
        out[(size_t)(rbase + i) * D_MODEL + col] =
            a[i] + bi + resid[(size_t)(rbase + i) * D_MODEL + col];
    }
  }
}

// ---------------------------------------------------------------- launch

extern "C" void kernel_launch(void* const* d_in, const int* in_sizes, int n_in,
                              void* d_out, int out_size, void* d_ws, size_t ws_size,
                              hipStream_t stream) {
  const float* x        = (const float*)d_in[0];   // [8,1024,1024]
  const float* ln_gamma = (const float*)d_in[1];   // [1024]
  const float* ln_beta  = (const float*)d_in[2];   // [1024]
  const float* in_w     = (const float*)d_in[3];   // [3072,1024]
  const float* in_b     = (const float*)d_in[4];   // [3072]
  const float* out_w    = (const float*)d_in[5];   // [1024,1024]
  const float* out_b    = (const float*)d_in[6];   // [1024]
  float* out            = (float*)d_out;

  f16* ws = (f16*)d_ws;
  f16* wqkv_h = ws;                                        // 3072*1024
  f16* wout_h = wqkv_h + (size_t)QKV_N * D_MODEL;          // 1024*1024
  f16* xn_h   = wout_h + (size_t)D_MODEL * D_MODEL;        // 8192*1024
  f16* qkv_h  = xn_h + (size_t)BS * D_MODEL;               // 8192*3072
  f16* ctx_h  = qkv_h + (size_t)BS * QKV_N;                // 8192*1024

  // 1) weight conversion to f16
  cvt_f32_to_f16<<<(QKV_N * D_MODEL) / 1024, 256, 0, stream>>>(in_w, wqkv_h,
                                                               (QKV_N * D_MODEL) / 4);
  cvt_f32_to_f16<<<(D_MODEL * D_MODEL) / 1024, 256, 0, stream>>>(out_w, wout_h,
                                                                 (D_MODEL * D_MODEL) / 4);
  // 2) LayerNorm
  ln_kernel<<<BS, 256, 0, stream>>>(x, ln_gamma, ln_beta, xn_h);
  // 3) QKV projection: 64 x 48 block tiles of 128x64
  gemm_qkv<<<(BS / 128) * (QKV_N / 64), 256, 0, stream>>>(xn_h, wqkv_h, in_b, qkv_h);
  // 4) flash attention: 8*16*(1024/128) = 1024 blocks
  attn_kernel<<<BATCH * HEADS * (SEQ / 128), 256, 0, stream>>>(qkv_h, ctx_h);
  // 5) out projection + residual: 64 x 16 block tiles
  gemm_out<<<(BS / 128) * (D_MODEL / 64), 256, 0, stream>>>(ctx_h, wout_h, out_b, x, out);
}
